// ParallelCausalSelfAttentionLayer_35270271435695
// MI455X (gfx1250) — matrix operand order repaired and checked by reading, not compiled, not run on hardware
//
#include <hip/hip_runtime.h>
#include <hip/hip_bf16.h>
#include <math.h>

// ---------------------------------------------------------------------------
// CDNA5 / gfx1250 wave32 WMMA attention pipeline, v3.1.
//  - bf16 matmul everywhere (fp32 accum), weights pre-transposed n-major
//  - double-buffered LDS tiles fed by global_load_async_to_lds_b128 (ASYNCcnt)
//  - attention K tile staged by the Tensor Data Mover (TENSORcnt)
// ---------------------------------------------------------------------------

typedef __attribute__((ext_vector_type(16))) __bf16 bf16x16;
typedef __attribute__((ext_vector_type(8)))  __bf16 bf16x8;
typedef __attribute__((ext_vector_type(2)))  __bf16 bf16x2;
typedef __attribute__((ext_vector_type(8)))  float  f32x8;
typedef __attribute__((ext_vector_type(4)))  int    i32x4;
typedef __attribute__((ext_vector_type(8)))  int    i32x8;
typedef __attribute__((ext_vector_type(4)))  unsigned int u32x4;

#define DEV_INLINE __device__ __forceinline__

constexpr int Bc   = 2;
constexpr int Sc   = 2048;
constexpr int Hc   = 2048;
constexpr int Dc   = 128;
constexpr int NQc  = 16;
constexpr int NKVc = 4;

typedef __attribute__((address_space(3))) void* as3_void_p;

#if defined(__has_builtin)
#if __has_builtin(__builtin_amdgcn_global_load_async_to_lds_b128) && \
    __has_builtin(__builtin_amdgcn_s_wait_asynccnt)
#define HAVE_ASYNC_LDS 1
#endif
#if __has_builtin(__builtin_amdgcn_tensor_load_to_lds) && \
    __has_builtin(__builtin_amdgcn_s_wait_tensorcnt)
#define HAVE_TDM 1
#endif
#endif
#ifndef HAVE_ASYNC_LDS
#define HAVE_ASYNC_LDS 0
#endif
#ifndef HAVE_TDM
#define HAVE_TDM 0
#endif

#if HAVE_ASYNC_LDS
typedef __attribute__((address_space(1))) i32x4* g_i32x4_p;
typedef __attribute__((address_space(3))) i32x4* s_i32x4_p;
DEV_INLINE void async_cp16(void* lds, const void* g) {
  __builtin_amdgcn_global_load_async_to_lds_b128((g_i32x4_p)g, (s_i32x4_p)lds,
                                                 0, 0);
}
DEV_INLINE void async_wait0() { __builtin_amdgcn_s_wait_asynccnt(0); }
DEV_INLINE void async_wait1() { __builtin_amdgcn_s_wait_asynccnt(1); }
#else
DEV_INLINE void async_cp16(void* lds, const void* g) {
  *(uint4*)lds = *(const uint4*)g;
}
DEV_INLINE void async_wait0() {}
DEV_INLINE void async_wait1() {}
#endif

#if HAVE_TDM
// 2D tile load via Tensor Data Mover. Element size = 2B (bf16).
// rowlen = tile/tensor dim0 in elements, rows = tile/tensor dim1,
// row_stride = tensor_dim0_stride in elements.
DEV_INLINE void tdm_load_2d_bf16(const void* gsrc, void* ldsdst, unsigned rows,
                                 unsigned rowlen, unsigned row_stride) {
  unsigned long long ga = (unsigned long long)(size_t)gsrc;
  unsigned lds = (unsigned)(size_t)(as3_void_p)ldsdst;
  u32x4 g0;
  g0[0] = 1u;                                          // count=1, flags=0
  g0[1] = lds;                                         // lds_addr
  g0[2] = (unsigned)(ga & 0xFFFFFFFFu);                // global_addr[31:0]
  g0[3] = (unsigned)((ga >> 32) & 0x01FFFFFFu) | (2u << 30);  // addr hi | type=2
  i32x8 g1;
  g1[0] = (int)(1u << 16);                             // data_size=1 (2 bytes)
  g1[1] = (int)((rowlen & 0xFFFFu) << 16);             // tensor_dim0[15:0]
  g1[2] = (int)((rowlen >> 16) | ((rows & 0xFFFFu) << 16));   // dim0 hi | dim1 lo
  g1[3] = (int)((rows >> 16) | ((rowlen & 0xFFFFu) << 16));   // dim1 hi | tile_dim0
  g1[4] = (int)(rows & 0xFFFFu);                       // tile_dim1 (tile_dim2=0)
  g1[5] = (int)row_stride;                             // tensor_dim0_stride[31:0]
  g1[6] = 0;
  g1[7] = 0;
  i32x4 z4 = {0, 0, 0, 0};
  i32x8 z8 = {0, 0, 0, 0, 0, 0, 0, 0};
  __builtin_amdgcn_tensor_load_to_lds(g0, g1, z4, z4, z8, 0);
}
#endif

DEV_INLINE f32x8 zero8() {
  f32x8 z;
#pragma unroll
  for (int i = 0; i < 8; ++i) z[i] = 0.0f;
  return z;
}

DEV_INLINE f32x8 wmma_bf16(bf16x16 a, bf16x16 b, f32x8 c) {
  return __builtin_amdgcn_wmma_f32_16x16x32_bf16(false, a, false, b,
                                                 (short)0, c, false, false);
}

// A-fragment (16x32): lanes 0-15 K={0..15}, lanes 16-31 K={16..31}.
DEV_INLINE bf16x16 frag_a(const __bf16* p) {
  bf16x8 lo = *(const bf16x8*)p;
  bf16x8 hi = *(const bf16x8*)(p + 8);
  return __builtin_shufflevector(lo, hi, 0, 1, 2, 3, 4, 5, 6, 7,
                                 8, 9, 10, 11, 12, 13, 14, 15);
}

// B-fragment (32x16): lane = column, 16 contiguous K (split at 16 by half-wave).
DEV_INLINE bf16x16 frag_b(const __bf16* p) { return *(const bf16x16*)p; }

// ---------------------------------------------------------------------------
__global__ void cvt_bf16_kernel(const float* __restrict__ src,
                                __bf16* __restrict__ dst, int n2) {
  int i = blockIdx.x * 256 + threadIdx.x;
  if (i < n2) {
    float2 v = ((const float2*)src)[i];
    bf16x2 o;
    o[0] = (__bf16)v.x;
    o[1] = (__bf16)v.y;
    ((bf16x2*)dst)[i] = o;
  }
}

// ---------------------------------------------------------------------------
__global__ __launch_bounds__(256)
void transpose_cvt_kernel(const float* __restrict__ W, __bf16* __restrict__ Wt,
                          int K, int N) {
  __shared__ float Ts[32][33];
  const int k0 = blockIdx.y * 32, n0 = blockIdx.x * 32;
  const int tid = threadIdx.x;
  for (int i = tid; i < 1024; i += 256) {
    int r = i >> 5, c = i & 31;
    Ts[r][c] = W[(size_t)(k0 + r) * N + n0 + c];
  }
  __syncthreads();
  for (int i = tid; i < 1024; i += 256) {
    int r = i >> 5, c = i & 31;
    Wt[(size_t)(n0 + r) * K + k0 + c] = (__bf16)Ts[c][r];
  }
}

// ---------------------------------------------------------------------------
// GEMM: C[M,N] = A[M,K] @ W + bias, A bf16 [M][K], Wt bf16 [N][K].
// Block = 8 waves, tile 128x64; double-buffered async staging + A prefetch.
// ---------------------------------------------------------------------------
__global__ __launch_bounds__(256)
void gemm_bf16(const __bf16* __restrict__ A, const __bf16* __restrict__ Wt,
               const float* __restrict__ bias, float* __restrict__ Cf,
               __bf16* __restrict__ Cb, int M, int N, int K) {
  __shared__ alignas(32) __bf16 Bs[2][64][32];   // ping-pong [n][k]

  const int tid   = threadIdx.x;
  const int lane  = tid & 31;
  const int wid   = tid >> 5;
  const int hi    = (lane >> 4) & 1;
  const int nbase = blockIdx.x * 64;
  const int mbase = blockIdx.y * 128 + wid * 16;

  f32x8 acc[4];
#pragma unroll
  for (int t = 0; t < 4; ++t) acc[t] = zero8();

  const int sn = tid >> 2, sc = tid & 3;
  const __bf16* wrow = Wt + (size_t)(nbase + sn) * K + sc * 8;
  const __bf16* arow = A + (size_t)(mbase + (lane & 15)) * K + hi * 16;

  const int T = K >> 5;
  async_cp16(&Bs[0][sn][sc * 8], wrow);
  bf16x16 af = frag_a(arow);
  int p = 0;

  for (int i = 0; i < T - 1; ++i) {
    async_cp16(&Bs[p ^ 1][sn][sc * 8], wrow + (i + 1) * 32);
    bf16x16 afn = frag_a(arow + (i + 1) * 32);
    async_wait1();
    __syncthreads();
#pragma unroll
    for (int t = 0; t < 4; ++t)
      acc[t] = wmma_bf16(af, frag_b(&Bs[p][t * 16 + (lane & 15)][hi * 16]), acc[t]);
    __syncthreads();
    af = afn;
    p ^= 1;
  }
  async_wait0();
  __syncthreads();
#pragma unroll
  for (int t = 0; t < 4; ++t)
    acc[t] = wmma_bf16(af, frag_b(&Bs[p][t * 16 + (lane & 15)][hi * 16]), acc[t]);

  const bool hasF = (Cf != nullptr), hasB = (Cb != nullptr);
#pragma unroll
  for (int t = 0; t < 4; ++t) {
    int n = nbase + t * 16 + (lane & 15);
    float bv = bias[n];
#pragma unroll
    for (int r = 0; r < 8; ++r) {
      int m = mbase + hi * 8 + r;
      float val = acc[t][r] + bv;
      if (hasF) Cf[(size_t)m * N + n] = val;
      if (hasB) Cb[(size_t)m * N + n] = (__bf16)val;
    }
  }
}

// ---------------------------------------------------------------------------
__global__ void rope_init_kernel(float* __restrict__ cs, float* __restrict__ sn) {
  int idx = blockIdx.x * 256 + threadIdx.x;   // S*64
  int d = idx & 63;
  int s = idx >> 6;
  float inv = powf(10000.0f, -(float)d * (1.0f / 64.0f));
  float f = (float)s * inv;
  cs[idx] = cosf(f);
  sn[idx] = sinf(f);
}

__global__ void rope_cvt_kernel(const float* __restrict__ src,
                                float* __restrict__ dstF,
                                __bf16* __restrict__ dstB,
                                const float* __restrict__ cs,
                                const float* __restrict__ sn, int nh) {
  int idx = blockIdx.x * 256 + threadIdx.x;   // rows*nh*64
  int d   = idx & 63;
  int h   = (idx >> 6) % nh;
  int row = idx / (nh * 64);
  int s   = row & (Sc - 1);
  size_t base = (size_t)row * (nh * Dc) + h * Dc + d;
  float x1 = src[base], x2 = src[base + 64];
  float c = cs[s * 64 + d], sv = sn[s * 64 + d];
  float o1 = x1 * c - x2 * sv;
  float o2 = x2 * c + x1 * sv;
  if (dstF) { dstF[base] = o1; dstF[base + 64] = o2; }
  dstB[base] = (__bf16)o1;
  dstB[base + 64] = (__bf16)o2;
}

// ---------------------------------------------------------------------------
// Flash attention: block = (128 q-rows, head, batch); 8 waves x 16 rows.
// Double-buffered tiles: K via TDM (or async-to-LDS), V transposed manually.
// ---------------------------------------------------------------------------
__global__ __launch_bounds__(256)
void attention_kernel(const __bf16* __restrict__ Q, const __bf16* __restrict__ Kb,
                      const __bf16* __restrict__ Vb, __bf16* __restrict__ Oattn) {
  __shared__ alignas(32) __bf16 Ks[2][32][128];   // [key][d]
  __shared__ alignas(32) __bf16 Vt[2][128][32];   // [d][key]
  __shared__ alignas(32) __bf16 Pb[8][512];       // per-wave P relayout scratch

  const int tid  = threadIdx.x;
  const int lane = tid & 31;
  const int wid  = tid >> 5;
  const int hi   = (lane >> 4) & 1;
  const int qt = blockIdx.x, h = blockIdx.y, b = blockIdx.z;
  const int kvh   = h >> 2;
  const int qbase = qt * 128;
  const int wrow  = qbase + wid * 16;
  const float scale = 0.08838834764831844f;    // 1/sqrt(128)
  const int kvstride = NKVc * Dc;              // 512

  bf16x16 qf[4];
  {
    const __bf16* qp =
        Q + (size_t)(b * Sc + wrow + (lane & 15)) * Hc + h * Dc + hi * 16;
#pragma unroll
    for (int c = 0; c < 4; ++c) qf[c] = frag_a(qp + c * 32);
  }

  f32x8 acc[8];
#pragma unroll
  for (int t = 0; t < 8; ++t) acc[t] = zero8();
  float mrun[8], lrun[8];
#pragma unroll
  for (int r = 0; r < 8; ++r) { mrun[r] = -3.0e38f; lrun[r] = 0.0f; }

  const __bf16* KbB = Kb + (size_t)b * Sc * kvstride + kvh * Dc;
  const unsigned short* VuB =
      (const unsigned short*)(Vb + (size_t)b * Sc * kvstride + kvh * Dc);
  __bf16* Pw = Pb[wid];

  auto stageK = [&](int kvb, int pp) {
#if HAVE_TDM
    if (wid == 0)
      tdm_load_2d_bf16(KbB + (size_t)kvb * kvstride, &Ks[pp][0][0], 32, 128,
                       (unsigned)kvstride);
#else
    int kk = tid >> 3, c = tid & 7;
    async_cp16(&Ks[pp][kk][c * 8], KbB + (size_t)(kvb + kk) * kvstride + c * 8);
#endif
  };
  auto stageV = [&](int kvb, int pp) {
    for (int i = tid; i < 128 * 16; i += 256) {
      int d = i & 127, kp = i >> 7;
      unsigned int v0 = VuB[(size_t)(kvb + 2 * kp) * kvstride + d];
      unsigned int v1 = VuB[(size_t)(kvb + 2 * kp + 1) * kvstride + d];
      *(unsigned int*)&Vt[pp][d][2 * kp] = v0 | (v1 << 16);
    }
  };
  auto waitK1 = [&]() {
#if HAVE_TDM
    if (wid == 0) __builtin_amdgcn_s_wait_tensorcnt(1);
#else
    async_wait1();
#endif
  };
  auto waitK0 = [&]() {
#if HAVE_TDM
    if (wid == 0) __builtin_amdgcn_s_wait_tensorcnt(0);
#else
    async_wait0();
#endif
  };

  const int T = (qbase + 128) >> 5;   // causal kv-tile count (>= 4)
  stageK(0, 0);
  stageV(0, 0);
  int p = 0;

  for (int i = 0; i < T; ++i) {
    const int kvb = i * 32;
    const bool more = (i + 1) < T;
    if (more) {
      stageK(kvb + 32, p ^ 1);
      stageV(kvb + 32, p ^ 1);
      waitK1();
    } else {
      waitK0();
    }
    __syncthreads();

    if (kvb < wrow + 16) {   // uniform per-wave causal skip
      f32x8 st0 = zero8(), st1 = zero8();
#pragma unroll
      for (int c = 0; c < 4; ++c) {
        st0 = wmma_bf16(qf[c], frag_b(&Ks[p][(lane & 15)][c * 32 + hi * 16]), st0);
        st1 = wmma_bf16(qf[c], frag_b(&Ks[p][16 + (lane & 15)][c * 32 + hi * 16]), st1);
      }

      float p0[8], p1[8];
#pragma unroll
      for (int r = 0; r < 8; ++r) {
        int m  = wrow + hi * 8 + r;
        int n0 = kvb + (lane & 15);
        float s0 = st0[r] * scale; if (n0 > m)      s0 = -3.0e38f;
        float s1 = st1[r] * scale; if (n0 + 16 > m) s1 = -3.0e38f;
        float rm = fmaxf(s0, s1);
#pragma unroll
        for (int off = 1; off < 16; off <<= 1) rm = fmaxf(rm, __shfl_xor(rm, off, 32));
        float newm = fmaxf(mrun[r], rm);
        float corr = __expf(mrun[r] - newm);
        float e0 = __expf(s0 - newm);
        float e1 = __expf(s1 - newm);
        float rs = e0 + e1;
#pragma unroll
        for (int off = 1; off < 16; off <<= 1) rs += __shfl_xor(rs, off, 32);
        lrun[r] = lrun[r] * corr + rs;
        mrun[r] = newm;
#pragma unroll
        for (int t = 0; t < 8; ++t) acc[t][r] *= corr;
        p0[r] = e0; p1[r] = e1;
      }

#pragma unroll
      for (int r = 0; r < 8; ++r) {
        int m = hi * 8 + r;
        Pw[m * 32 + (lane & 15)]      = (__bf16)p0[r];
        Pw[m * 32 + (lane & 15) + 16] = (__bf16)p1[r];
      }
      bf16x16 pa = frag_a(&Pw[(lane & 15) * 32 + hi * 16]);

#pragma unroll
      for (int t = 0; t < 8; ++t) {
        acc[t] = wmma_bf16(pa, frag_b(&Vt[p][t * 16 + (lane & 15)][hi * 16]), acc[t]);
      }
    }
    __syncthreads();
    p ^= 1;
  }

#pragma unroll
  for (int r = 0; r < 8; ++r) {
    float inv = 1.0f / lrun[r];
    int m = wrow + hi * 8 + r;
    __bf16* orow = Oattn + (size_t)(b * Sc + m) * Hc + h * Dc;
#pragma unroll
    for (int t = 0; t < 8; ++t)
      orow[t * 16 + (lane & 15)] = (__bf16)(acc[t][r] * inv);
  }
}

// ---------------------------------------------------------------------------
extern "C" void kernel_launch(void* const* d_in, const int* in_sizes, int n_in,
                              void* d_out, int out_size, void* d_ws, size_t ws_size,
                              hipStream_t stream) {
  (void)in_sizes; (void)n_in; (void)out_size; (void)ws_size;

  const float* hidden = (const float*)d_in[0];
  const float* Wq = (const float*)d_in[1];
  const float* bq = (const float*)d_in[2];
  const float* Wk = (const float*)d_in[3];
  const float* bk = (const float*)d_in[4];
  const float* Wv = (const float*)d_in[5];
  const float* bv = (const float*)d_in[6];
  const float* Wc = (const float*)d_in[7];
  const float* bc = (const float*)d_in[8];

  float* out  = (float*)d_out;
  float* kout = out  + (size_t)Bc * Sc * NQc  * Dc;
  float* vout = kout + (size_t)Bc * Sc * NKVc * Dc;

  const int rows = Bc * Sc;      // 4096
  const int NKV  = NKVc * Dc;    // 512

  char* w = (char*)d_ws;
  auto alloc = [&](size_t bytes) -> char* {
    char* p = w;
    w += (bytes + 255) & ~(size_t)255;
    return p;
  };
  __bf16* hiddenB = (__bf16*)alloc((size_t)rows * Hc * 2);
  __bf16* WqT     = (__bf16*)alloc((size_t)Hc * Hc * 2);
  __bf16* WkT     = (__bf16*)alloc((size_t)Hc * NKV * 2);
  __bf16* WvT     = (__bf16*)alloc((size_t)Hc * NKV * 2);
  __bf16* WcT     = (__bf16*)alloc((size_t)Hc * Hc * 2);
  float*  Qws     = (float*)alloc((size_t)rows * Hc * 4);
  __bf16* Qb      = (__bf16*)alloc((size_t)rows * Hc * 2);
  __bf16* Kbb     = (__bf16*)alloc((size_t)rows * NKV * 2);
  __bf16* Vbb     = (__bf16*)alloc((size_t)rows * NKV * 2);
  __bf16* attnB   = (__bf16*)alloc((size_t)rows * Hc * 2);
  float*  cs      = (float*)alloc((size_t)Sc * 64 * 4);
  float*  sn      = (float*)alloc((size_t)Sc * 64 * 4);

  cvt_bf16_kernel<<<(rows * Hc / 2) / 256, 256, 0, stream>>>(hidden, hiddenB,
                                                             rows * Hc / 2);
  transpose_cvt_kernel<<<dim3(Hc / 32, Hc / 32), 256, 0, stream>>>(Wq, WqT, Hc, Hc);
  transpose_cvt_kernel<<<dim3(NKV / 32, Hc / 32), 256, 0, stream>>>(Wk, WkT, Hc, NKV);
  transpose_cvt_kernel<<<dim3(NKV / 32, Hc / 32), 256, 0, stream>>>(Wv, WvT, Hc, NKV);
  transpose_cvt_kernel<<<dim3(Hc / 32, Hc / 32), 256, 0, stream>>>(Wc, WcT, Hc, Hc);
  rope_init_kernel<<<(Sc * 64) / 256, 256, 0, stream>>>(cs, sn);

  gemm_bf16<<<dim3(Hc / 64, rows / 128), 256, 0, stream>>>(
      hiddenB, WqT, bq, Qws, nullptr, rows, Hc, Hc);
  gemm_bf16<<<dim3(NKV / 64, rows / 128), 256, 0, stream>>>(
      hiddenB, WkT, bk, kout, nullptr, rows, NKV, Hc);
  gemm_bf16<<<dim3(NKV / 64, rows / 128), 256, 0, stream>>>(
      hiddenB, WvT, bv, vout, Vbb, rows, NKV, Hc);

  rope_cvt_kernel<<<(rows * NQc * 64) / 256, 256, 0, stream>>>(
      Qws, nullptr, Qb, cs, sn, NQc);
  rope_cvt_kernel<<<(rows * NKVc * 64) / 256, 256, 0, stream>>>(
      kout, kout, Kbb, cs, sn, NKVc);

  attention_kernel<<<dim3(Sc / 128, NQc, Bc), 256, 0, stream>>>(Qb, Kbb, Vbb, attnB);

  gemm_bf16<<<dim3(Hc / 64, rows / 128), 256, 0, stream>>>(
      attnB, WcT, bc, out, nullptr, rows, Hc, Hc);
}
